// TiltedStableSDEFinancial_75737453298061
// MI455X (gfx1250) — matrix operand order repaired and checked by reading, not compiled, not run on hardware
//
#include <hip/hip_runtime.h>
#include <hip/hip_bf16.h>

typedef __attribute__((ext_vector_type(16))) _Float16 v16h;
typedef __attribute__((ext_vector_type(8)))  float    v8f;

#define Bn 32
#define Tn 64
#define Dn 32
#define Sn 64
#define Jn 16
#define Rn 8
#define Wn 64
#define PWn 64

#define TAUc      0.1f
#define INV_ALPHA 0.83333333333f      /* 1/1.2 */
#define SIG2      0.0025f             /* 0.05^2 */
#define RATE1c    13.2074432f         /* (1/1.2)*0.1^-1.2 */
#define MAXE      80.0f
#define EPSF      1e-6f
#define CLIPV     2.5f                /* 50*sigma */

// ---- fast transcendentals: map to single CDNA5 TRANS32 ops where possible ----
__device__ __forceinline__ float ftanh(float x) {
#if __has_builtin(__builtin_amdgcn_tanhf)
  return __builtin_amdgcn_tanhf(x);          // v_tanh_f32
#else
  return tanhf(x);
#endif
}
__device__ __forceinline__ float fexp(float x)  { return __expf(x); }   // v_exp_f32
__device__ __forceinline__ float flog(float x)  { return __logf(x); }   // v_log_f32
__device__ __forceinline__ float fpow(float a, float b) { return __powf(a, b); }
__device__ __forceinline__ float fcos(float x)  { return __cosf(x); }   // v_cos_f32

__device__ __forceinline__ uint32_t mix32(uint32_t x) {
  x ^= x >> 16; x *= 0x7feb352du;
  x ^= x >> 15; x *= 0x846ca68bu;
  x ^= x >> 16; return x;
}
// deterministic counter-based uniform in (0,1); stream = (step, phase, element, sub)
__device__ __forceinline__ float rng01(uint32_t st, uint32_t ph, uint32_t eid, uint32_t sub) {
  uint32_t a = st * 0x9E3779B9u + ph * 0x7F4A7C15u + eid * 0x94D049BBu;
  uint32_t h = mix32(a ^ (sub * 0x85EBCA6Bu + 0xC2B2AE35u));
  h = mix32(h + a);
  return ((float)(h >> 8) + 0.5f) * (1.0f / 16777216.0f);
}

__device__ __forceinline__ float softplusf(float x) {
  return (x > 20.0f) ? x : log1pf(fexp(x));
}

// A-fragment: 16x32 f16 tile (rows m0..m0+15, cols k0..k0+31) from row-major [.., ld]
// ISA layout: lanes 0-15 / 16-31 split K into +0 / +8 (and +16/+24 in VGPRs 4-7)
__device__ __forceinline__ v16h frag_a(const _Float16* Ab, int m0, int ld, int k0, int lane) {
  const int m  = m0 + (lane & 15);
  const int hi = (lane >> 4) & 1;
  const _Float16* r0 = Ab + m * ld + k0 + 8 * hi;
  const _Float16* r1 = r0 + 16;
  v16h a;
#pragma unroll
  for (int i = 0; i < 4; ++i) {
    a[2*i]     = r0[2*i];
    a[2*i + 1] = r0[2*i + 1];
    a[8 + 2*i]     = r1[2*i];
    a[8 + 2*i + 1] = r1[2*i + 1];
  }
  return a;
}

// B-fragment: 32x16 f16 tile (rows k0..k0+31, cols n0..n0+15), B row-major [K, ld]
// lanes 0-15 / 16-31 split K into +0 / +8 (and +16/+24 in elements 8-15)
__device__ __forceinline__ v16h frag_b(const _Float16* Bb, int n0, int ld, int k0, int lane) {
  const int n  = n0 + (lane & 15);
  const int hi = (lane >> 4) & 1;
  const _Float16* base = Bb + (k0 + 8 * hi) * ld + n;
  v16h bv;
#pragma unroll
  for (int i = 0; i < 4; ++i) {
    bv[2*i]     = base[(2*i) * ld];
    bv[2*i + 1] = base[(2*i + 1) * ld];
    bv[8 + 2*i]     = base[(16 + 2*i) * ld];
    bv[8 + 2*i + 1] = base[(16 + 2*i + 1) * ld];
  }
  return bv;
}

__global__ __launch_bounds__(1024)
void tilted_stable_sde_kernel(
    const float* __restrict__ X0,  const float* __restrict__ ts,
    const float* __restrict__ dW1, const float* __restrict__ db1,
    const float* __restrict__ dW2, const float* __restrict__ db2,
    const float* __restrict__ dW3, const float* __restrict__ db3,
    const float* __restrict__ dlog,
    const float* __restrict__ pW1, const float* __restrict__ pb1,
    const float* __restrict__ pW2, const float* __restrict__ pb2,
    const float* __restrict__ pW3, const float* __restrict__ pb3,
    float* __restrict__ out)
{
  __shared__ float    Xs[Bn * Dn];            // current state, f32
  __shared__ _Float16 Xh[Bn * Dn];            // WMMA A staging
  __shared__ _Float16 W1h[(Dn + 1) * Wn];     // drift weights, f16
  __shared__ _Float16 W2h[Wn * Wn];
  __shared__ _Float16 W3h[Wn * Dn];
  __shared__ _Float16 H1h[Bn * Wn];
  __shared__ _Float16 H2h[Bn * Wn];
  __shared__ float    OUTs[Bn * Dn];
  __shared__ float    As[Dn], Bs[Dn], Dsc[Dn];
  __shared__ float    ph1[PWn], ph2[PWn], pout[2 * Dn];

  const int tid  = threadIdx.x;       // 0..1023 == b*32 + d
  const int lane = tid & 31;
  const int wv   = tid >> 5;          // wave id 0..31
  const int d    = tid & 31;

  if (tid == 0) {                     // gfx1250 global_prefetch_b8 into L2
    __builtin_prefetch(dW1, 0, 3);
    __builtin_prefetch(dW2, 0, 3);
    __builtin_prefetch(dW3, 0, 3);
    __builtin_prefetch(pW2, 0, 3);
    __builtin_prefetch(pW3, 0, 3);
  }

  // one-time init: state + f16 weight images in LDS
  Xs[tid] = X0[tid];
  for (int i = tid; i < (Dn + 1) * Wn; i += 1024) W1h[i] = (_Float16)dW1[i];
  for (int i = tid; i < Wn * Wn;       i += 1024) W2h[i] = (_Float16)dW2[i];
  for (int i = tid; i < Wn * Dn;       i += 1024) W3h[i] = (_Float16)dW3[i];
  if (tid < Dn) Dsc[tid] = softplusf(dlog[tid]);
  __syncthreads();

  for (int st = 0; st < Tn; ++st) {
    const float t  = ts[st];
    const float tp = ts[st > 0 ? st - 1 : 0];
    const float dt = t - tp;

    // ---- phi MLP (coefficients A_t, B_t) + stage X as f16 ----
    if (tid < PWn) ph1[tid] = ftanh(t * pW1[tid] + pb1[tid]);
    Xh[tid] = (_Float16)Xs[tid];
    __syncthreads();
    if (tid < PWn) {
      float s = pb2[tid];
#pragma unroll 8
      for (int k = 0; k < PWn; ++k) s += ph1[k] * pW2[k * PWn + tid];
      ph2[tid] = ftanh(s);
    }
    __syncthreads();
    if (tid < 2 * Dn) {
      float s = pb3[tid];
#pragma unroll 8
      for (int k = 0; k < PWn; ++k) s += ph2[k] * pW3[k * 2 * Dn + tid];
      pout[tid] = s;
    }
    __syncthreads();
    if (tid < Dn) { As[tid] = -softplusf(pout[tid]); Bs[tid] = pout[Dn + tid]; }
    __syncthreads();

    const float A  = As[d];
    const float Bt = Bs[d];
    const float ds = Dsc[d];
    float Xv = Xs[tid];
    const float K1 = 2.0f * A * Xv + Bt;

    // ---- tilt weights: pass 1 = top-5 (for 0.95 quantile of 64 samples) ----
    float t0 = -1e30f, t1 = -1e30f, t2 = -1e30f, t3 = -1e30f, t4 = -1e30f;
#pragma unroll 4
    for (int s = 0; s < Sn; ++s) {
      const float u    = rng01(st, 0, tid, s);
      const float raw  = TAUc * fpow(1.0f - u, -INV_ALPHA);
      const float sc   = ds * raw;
      const float quad = A * sc * sc;
      const float lin  = K1 * sc;
      const float w    = fexp(MAXE * ftanh((quad + lin) * (1.0f / MAXE)))
                       + fexp(MAXE * ftanh((quad - lin) * (1.0f / MAXE)));
      if (w > t4) {
        t4 = w;
        if (t4 > t3) { float tm = t3; t3 = t4; t4 = tm; }
        if (t3 > t2) { float tm = t2; t2 = t3; t3 = tm; }
        if (t2 > t1) { float tm = t1; t1 = t2; t2 = tm; }
        if (t1 > t0) { float tm = t0; t0 = t1; t1 = tm; }
      }
    }
    // sorted[59] = 5th largest = t4, sorted[60] = 4th largest = t3; q = 0.95*63 = 59.85
    const float tq = 0.15f * t4 + 0.85f * t3;

    // ---- pass 2: regenerate, capped mean -> Poisson rate ----
    float accw = 0.0f;
#pragma unroll 4
    for (int s = 0; s < Sn; ++s) {
      const float u    = rng01(st, 0, tid, s);
      const float raw  = TAUc * fpow(1.0f - u, -INV_ALPHA);
      const float sc   = ds * raw;
      const float quad = A * sc * sc;
      const float lin  = K1 * sc;
      const float w    = fexp(MAXE * ftanh((quad + lin) * (1.0f / MAXE)))
                       + fexp(MAXE * ftanh((quad - lin) * (1.0f / MAXE)));
      accw += fminf(w, tq);
    }
    const float lam = (accw * (1.0f / (float)Sn)) * RATE1c * dt;

    // ---- Poisson (Knuth) ----
    const float L = fexp(-lam);
    float pp = 1.0f; int kk = 0;
    do { pp *= rng01(st, 1, tid, kk); ++kk; } while (pp > L && kk < 80);
    const int njump = kk - 1;

    // ---- rejection-sampled mixing variable + conditional Gaussian ----
    const float Mb = fexp(-K1 * K1 / (4.0f * A));
    float soj = 0.0f;
    for (int j = 0; j < Jn; ++j) {
      float r = TAUc; bool got = false;
#pragma unroll
      for (int at = 0; at < Rn; ++at) {
        const uint32_t sub = (uint32_t)(j * Rn + at);
        const float u   = rng01(st, 2, tid, sub);
        const float rp  = TAUc * fpow(1.0f - u, -INV_ALPHA);
        const float rss = rp * rp * SIG2;
        const float den = fmaxf(1.0f - 2.0f * A * rss, EPSF);
        const float C   = fexp(K1 * K1 * rss / (2.0f * den)) * __frsqrt_rn(den);
        const float pa  = fminf(C / Mb, 1.0f);
        const float ua  = rng01(st, 3, tid, sub);
        if (!got && ua < pa) { r = rp; got = true; }
      }
      const float K2  = A - 1.0f / (2.0f * r * r * SIG2);
      const float K2s = fminf(K2, -EPSF);
      const float mean = -K1 / (2.0f * K2s);
      const float var  = -1.0f / (2.0f * K2s);
      const float u1 = rng01(st, 4, tid, j);
      const float u2 = rng01(st, 5, tid, j);
      const float z  = sqrtf(-2.0f * flog(u1)) * fcos(6.28318530718f * u2);
      const float cond = mean + sqrtf(var) * z;
      if (j < njump) soj += cond;
    }
    soj = CLIPV * ftanh(soj * (1.0f / CLIPV));

    // ---- drift MLP via WMMA: layer 1, H1 = tanh([t,X] @ dW1 + b1) ----
    if (wv < 8) {
      const int m0 = (wv >> 2) * 16, n0 = (wv & 3) * 16;
      v16h a  = frag_a(Xh, m0, Dn, 0, lane);
      v16h bf = frag_b(W1h + Wn, n0, Wn, 0, lane);  // skip t-row (row 0)
      v8f c = {};
      c = __builtin_amdgcn_wmma_f32_16x16x32_f16(false, a, false, bf, (short)0, c, false, false);
      const int hi = (lane >> 4) & 1, col = n0 + (lane & 15);
#pragma unroll
      for (int v = 0; v < 8; ++v) {
        const int row = m0 + v + 8 * hi;
        const float val = c[v] + t * dW1[col] + db1[col];   // fold t-column + bias
        H1h[row * Wn + col] = (_Float16)ftanh(val);
      }
    }
    __syncthreads();

    // ---- layer 2: H2 = tanh(H1 @ dW2 + b2), K = 64 ----
    if (wv < 8) {
      const int m0 = (wv >> 2) * 16, n0 = (wv & 3) * 16;
      v8f c = {};
#pragma unroll
      for (int k0 = 0; k0 < Wn; k0 += 32) {
        v16h a  = frag_a(H1h, m0, Wn, k0, lane);
        v16h bf = frag_b(W2h, n0, Wn, k0, lane);
        c = __builtin_amdgcn_wmma_f32_16x16x32_f16(false, a, false, bf, (short)0, c, false, false);
      }
      const int hi = (lane >> 4) & 1, col = n0 + (lane & 15);
#pragma unroll
      for (int v = 0; v < 8; ++v) {
        const int row = m0 + v + 8 * hi;
        H2h[row * Wn + col] = (_Float16)ftanh(c[v] + db2[col]);
      }
    }
    __syncthreads();

    // ---- layer 3: drift = H2 @ dW3 + b3, N = 32 ----
    if (wv < 4) {
      const int m0 = (wv >> 1) * 16, n0 = (wv & 1) * 16;
      v8f c = {};
#pragma unroll
      for (int k0 = 0; k0 < Wn; k0 += 32) {
        v16h a  = frag_a(H2h, m0, Wn, k0, lane);
        v16h bf = frag_b(W3h, n0, Dn, k0, lane);
        c = __builtin_amdgcn_wmma_f32_16x16x32_f16(false, a, false, bf, (short)0, c, false, false);
      }
      const int hi = (lane >> 4) & 1, col = n0 + (lane & 15);
#pragma unroll
      for (int v = 0; v < 8; ++v) {
        const int row = m0 + v + 8 * hi;
        OUTs[row * Dn + col] = c[v] + db3[col];
      }
    }
    __syncthreads();

    // ---- state update + emit path[st] ----
    Xv = Xv + OUTs[tid] * dt + ds * soj;
    Xs[tid] = Xv;
    out[st * (Bn * Dn) + tid] = Xv;
    __syncthreads();
  }
}

extern "C" void kernel_launch(void* const* d_in, const int* in_sizes, int n_in,
                              void* d_out, int out_size, void* d_ws, size_t ws_size,
                              hipStream_t stream) {
  (void)in_sizes; (void)n_in; (void)out_size; (void)d_ws; (void)ws_size;
  tilted_stable_sde_kernel<<<1, 1024, 0, stream>>>(
      (const float*)d_in[0],  (const float*)d_in[1],  (const float*)d_in[2],
      (const float*)d_in[3],  (const float*)d_in[4],  (const float*)d_in[5],
      (const float*)d_in[6],  (const float*)d_in[7],  (const float*)d_in[8],
      (const float*)d_in[9],  (const float*)d_in[10], (const float*)d_in[11],
      (const float*)d_in[12], (const float*)d_in[13], (const float*)d_in[14],
      (float*)d_out);
}
